// FutureSceneDecoder_69209103008094
// MI455X (gfx1250) — matrix operand order repaired and checked by reading, not compiled, not run on hardware
//
#include <hip/hip_runtime.h>
#include <hip/hip_bf16.h>
#include <math.h>

// ---------------- types for WMMA ----------------
typedef __attribute__((ext_vector_type(16))) __bf16     v16bf;
typedef __attribute__((ext_vector_type(8)))  float      v8f;
typedef __attribute__((ext_vector_type(4)))  unsigned   u32x4;

union A16 {
    v16bf          v;
    u32x4          q[2];
    unsigned int   w[8];
    unsigned short u[16];
};

// native f32 -> bf16 (RTNE) conversion; backend emits the hardware cvt
static __device__ __forceinline__ unsigned short f2bf(float f) {
    union { __bf16 h; unsigned short u; } x;
    x.h = (__bf16)f;
    return x.u;
}
static __device__ __forceinline__ float bf2f(unsigned short h) {
    union { unsigned u; float f; } x; x.u = ((unsigned)h) << 16;
    return x.f;
}
// tanh(x) = 1 - 2/(e^{2x}+1), via hardware exp2/rcp (saturates correctly)
static __device__ __forceinline__ float tanh_fast(float x) {
    float e = __builtin_amdgcn_exp2f(x * 2.8853900817779268f); // 2*log2(e)
    return 1.0f - 2.0f * __builtin_amdgcn_rcpf(e + 1.0f);
}
static __device__ __forceinline__ v8f splat8(float x) {
    v8f r;
#pragma unroll
    for (int k = 0; k < 8; ++k) r[k] = x;
    return r;
}
static __device__ __forceinline__ v8f wmma_bf(const A16& a, const A16& b, v8f c) {
    return __builtin_amdgcn_wmma_f32_16x16x32_bf16(false, a.v, false, b.v,
                                                   (short)0, c, false, false);
}

// ---------------- problem constants ----------------
#define NB   256     // graphs
#define NN   32      // nodes per graph
#define EMB  128
#define XS   232     // x stride (halves): 116 dwords -> bank rotation per row
#define HB   136     // h stride (halves): 68 dwords -> conflict-free A loads
#define HS   136     // per-wave hidden tile stride
#define US   264     // u=[h,aggr] stride
// workspace offsets (bf16 halves)
#define OFF_LIN   0u
#define OFF_MSG1  28672u
#define OFF_MSG2  176128u
#define OFF_UPD1  241664u
#define OFF_UPD2  372736u

// ---------------- weight pre-pack into WMMA B-operand layout ----------------
__global__ __launch_bounds__(256) void pack_b(const float* __restrict__ W,
                                              unsigned short* __restrict__ dst,
                                              int OUT, int IN, int KC) {
    long idx  = (long)blockIdx.x * 256 + threadIdx.x;
    int  e    = (int)(idx & 15);
    int  lane = (int)((idx >> 4) & 31);
    int  nc   = (int)((idx >> 9) & 7);
    long rest = idx >> 12;
    int  kc   = (int)(rest % KC);
    int  l    = (int)(rest / KC);
    int  n    = nc * 16 + (lane & 15);
    int  k    = kc * 32 + ((lane >> 4) << 3) + (e & 7) + ((e >> 3) << 4);
    float v = (k < IN) ? W[(long)l * OUT * IN + (long)n * IN + k] : 0.0f;
    dst[idx] = f2bf(v);
}

// ---------------- main GNN kernel: one workgroup per graph ----------------
__global__ __launch_bounds__(256, 1) void gnn_main(
    const float* __restrict__ pos,       const float* __restrict__ enc,
    const float* __restrict__ pos_emb,   const float* __restrict__ na_emb,
    const int*   __restrict__ Tcls,
    const float* __restrict__ fc1_w,     const float* __restrict__ fc1_b,
    const float* __restrict__ fc2_w,     const float* __restrict__ fc2_b,
    const float* __restrict__ lin_b,
    const float* __restrict__ msg_b1,    const float* __restrict__ msg_b2,
    const float* __restrict__ upd_b1,    const float* __restrict__ upd_b2,
    const unsigned short* __restrict__ wsb,
    float* __restrict__ out)
{
    __shared__ unsigned short scratch[17408];      // xbf(32*232) / hid(8*16*136) / ubf+h1b
    __shared__ unsigned short hbf[NN * HB];        // node embeddings (bf16), padded stride
    __shared__ float          aggr[NN * EMB];
    __shared__ float          e1[64];
    __shared__ float          evec[EMB];
    __shared__ float          pos_s[NN * 2];
    __shared__ unsigned short toh[NN * 2];

    const int b    = blockIdx.x;
    const int tid  = threadIdx.x;
    const int lane = tid & 31;
    const int wv   = tid >> 5;
    const int hi8  = (lane >> 4) << 3;   // A-operand K-half offset
    const int mhi  = (lane >> 4) << 3;   // C-operand M-half offset
    const int ncol = lane & 15;

    // ---- decoder_fc: e = tanh(enc @ fc1^T + b1) @ fc2^T + b2 ----
    if (tid < 64) {
        float s = fc1_b[tid];
        const float* er = enc + (size_t)b * 256;
        const float* wr = fc1_w + (size_t)tid * 256;
        for (int k = 0; k < 256; ++k) s += er[k] * wr[k];
        e1[tid] = tanh_fast(s);
    }
    __syncthreads();
    if (tid < 128) {
        float s = fc2_b[tid];
        const float* wr = fc2_w + (size_t)tid * 64;
        for (int k = 0; k < 64; ++k) s += e1[k] * wr[k];
        evec[tid] = s;
    }
    if (tid < NN) {
        pos_s[tid * 2 + 0] = pos[((size_t)b * NN + tid) * 2 + 0];
        pos_s[tid * 2 + 1] = pos[((size_t)b * NN + tid) * 2 + 1];
        int tv = Tcls[(size_t)b * NN + tid];
        toh[tid * 2 + 0] = f2bf(tv == 0 ? 1.0f : 0.0f);
        toh[tid * 2 + 1] = f2bf(tv == 1 ? 1.0f : 0.0f);
    }
    __syncthreads();

    // ---- build x (bf16, K padded to 224 zeros, stride 232) ----
    {
        unsigned short* xbf = scratch;
        float na = na_emb[b];
        for (int idx = tid; idx < NN * XS; idx += 256) {
            int i = idx / XS, k = idx % XS;
            float v;
            if      (k < 64)  v = pos_emb[((size_t)b * NN + i) * 64 + k];
            else if (k == 64) v = na;
            else if (k < 193) v = evec[k - 65];
            else              v = 0.0f;
            xbf[idx] = f2bf(v);
        }
    }
    __syncthreads();

    // ---- h0 = x @ lin_w^T + lin_b  (M=32, K=224, N=128) ----
    {
        const unsigned short* wp  = wsb + OFF_LIN;
        const unsigned short* xbf = scratch;
        int rt  = wv & 1;
        int nc0 = (wv >> 1) * 2;
        v8f c[2];
#pragma unroll
        for (int s = 0; s < 2; ++s) c[s] = splat8(lin_b[(nc0 + s) * 16 + ncol]);
        int arow = rt * 16 + (lane & 15);
#pragma unroll 1
        for (int kc = 0; kc < 7; ++kc) {
            A16 a;
            const u32x4* p = (const u32x4*)&xbf[arow * XS + kc * 32 + hi8];
            a.q[0] = p[0]; a.q[1] = p[2];
            const u32x4* wq = (const u32x4*)(wp + (size_t)kc * 8 * 512);
#pragma unroll
            for (int s = 0; s < 2; ++s) {
                A16 bb;
                bb.q[0] = wq[(nc0 + s) * 64 + lane * 2];
                bb.q[1] = wq[(nc0 + s) * 64 + lane * 2 + 1];
                c[s] = wmma_bf(a, bb, c[s]);
            }
        }
#pragma unroll
        for (int s = 0; s < 2; ++s)
#pragma unroll
            for (int r = 0; r < 8; ++r)
                hbf[(rt * 16 + r + mhi) * HB + (nc0 + s) * 16 + ncol] = f2bf(c[s][r]);
    }

    // ---- L = 4 message-passing layers ----
#pragma unroll 1
    for (int l = 0; l < 4; ++l) {
        for (int idx = tid; idx < NN * EMB; idx += 256) aggr[idx] = 0.0f;
        __syncthreads();

        const unsigned short* w1p = wsb + OFF_MSG1 + (size_t)l * 9 * 4096;
        const unsigned short* w2p = wsb + OFF_MSG2 + (size_t)l * 4 * 4096;
        const float* b1 = msg_b1 + l * EMB;
        const float* b2 = msg_b2 + l * EMB;
        unsigned short* hid = scratch + wv * (16 * HS);   // wave-private 16x128

        // 64 edge-row tiles (src j fixed per tile, dst i-half fixed), 8 per wave
#pragma unroll 1
        for (int t = wv; t < 64; t += 8) {
            int j     = t >> 1;
            int ibase = (t & 1) * 16;
            int arow  = ibase + (lane & 15);

            // GEMM1: (16 x 288) @ (288 x 128), N split in 2 groups of 4 tiles
#pragma unroll 1
            for (int g = 0; g < 2; ++g) {
                v8f c[4];
#pragma unroll
                for (int nc = 0; nc < 4; ++nc)
                    c[nc] = splat8(b1[(g * 4 + nc) * 16 + ncol]);
#pragma unroll 1
                for (int kc = 0; kc < 9; ++kc) {
                    A16 a;
                    if (kc < 4) {                       // h[dst=i]
                        const u32x4* p = (const u32x4*)&hbf[arow * HB + kc * 32 + hi8];
                        a.q[0] = p[0]; a.q[1] = p[2];
                    } else if (kc < 8) {                // h[src=j] (row-uniform)
                        const u32x4* p = (const u32x4*)&hbf[j * HB + (kc - 4) * 32 + hi8];
                        a.q[0] = p[0]; a.q[1] = p[2];
                    } else {                            // edge_attr: [oh(j), oh(i), dist]
#pragma unroll
                        for (int z = 0; z < 8; ++z) a.w[z] = 0u;
                        if (lane < 16) {
                            int i = ibase + lane;
                            a.u[0] = toh[j * 2 + 0];
                            a.u[1] = toh[j * 2 + 1];
                            a.u[2] = toh[i * 2 + 0];
                            a.u[3] = toh[i * 2 + 1];
                            float dx = pos_s[j * 2] - pos_s[i * 2];
                            float dy = pos_s[j * 2 + 1] - pos_s[i * 2 + 1];
                            a.u[4] = f2bf(sqrtf(dx * dx + dy * dy));
                        }
                    }
                    const u32x4* wq = (const u32x4*)(w1p + (size_t)kc * 8 * 512);
#pragma unroll
                    for (int nc = 0; nc < 4; ++nc) {
                        A16 bb;
                        bb.q[0] = wq[(g * 4 + nc) * 64 + lane * 2];
                        bb.q[1] = wq[(g * 4 + nc) * 64 + lane * 2 + 1];
                        c[nc] = wmma_bf(a, bb, c[nc]);
                    }
                }
                // tanh -> wave-private LDS tile (A operand for GEMM2)
#pragma unroll
                for (int nc = 0; nc < 4; ++nc)
#pragma unroll
                    for (int r = 0; r < 8; ++r)
                        hid[(r + mhi) * HS + (g * 4 + nc) * 16 + ncol] =
                            f2bf(tanh_fast(c[nc][r]));
            }

            // GEMM2: (16 x 128) @ (128 x 128), N split in 2 groups of 4
#pragma unroll 1
            for (int g = 0; g < 2; ++g) {
                v8f c2[4];
#pragma unroll
                for (int nc = 0; nc < 4; ++nc)
                    c2[nc] = splat8(b2[(g * 4 + nc) * 16 + ncol]);
#pragma unroll 1
                for (int kc = 0; kc < 4; ++kc) {
                    A16 a;
                    const u32x4* p = (const u32x4*)&hid[(lane & 15) * HS + kc * 32 + hi8];
                    a.q[0] = p[0]; a.q[1] = p[2];
                    const u32x4* wq = (const u32x4*)(w2p + (size_t)kc * 8 * 512);
#pragma unroll
                    for (int nc = 0; nc < 4; ++nc) {
                        A16 bb;
                        bb.q[0] = wq[(g * 4 + nc) * 64 + lane * 2];
                        bb.q[1] = wq[(g * 4 + nc) * 64 + lane * 2 + 1];
                        c2[nc] = wmma_bf(a, bb, c2[nc]);
                    }
                }
                // tanh + scatter-add into aggr[dst=i]
#pragma unroll
                for (int nc = 0; nc < 4; ++nc)
#pragma unroll
                    for (int r = 0; r < 8; ++r)
                        atomicAdd(&aggr[(ibase + r + mhi) * EMB + (g * 4 + nc) * 16 + ncol],
                                  tanh_fast(c2[nc][r]));
            }
        }
        __syncthreads();

        // ---- update MLP: u = [h, aggr] (32 x 256, stride 264) ----
        unsigned short* ubf = scratch;             // 32*264 = 8448 halves
        unsigned short* h1b = scratch + 8448;      // 32*136 = 4352 halves
        for (int idx = tid; idx < NN * US; idx += 256) {
            int i = idx / US, k = idx % US;
            unsigned short v;
            if      (k < EMB)  v = hbf[i * HB + k];
            else if (k < 256)  v = f2bf(aggr[i * EMB + (k - EMB)]);
            else               v = 0;
            ubf[idx] = v;
        }
        __syncthreads();

        const unsigned short* u1p = wsb + OFF_UPD1 + (size_t)l * 8 * 4096;
        const unsigned short* u2p = wsb + OFF_UPD2 + (size_t)l * 4 * 4096;
        const float* ub1 = upd_b1 + l * EMB;
        const float* ub2 = upd_b2 + l * EMB;
        int rt   = wv & 1;
        int nc0  = (wv >> 1) * 2;
        int urow = rt * 16 + (lane & 15);

        // GEMM1: (32 x 256) @ (256 x 128)
        v8f cu[2];
#pragma unroll
        for (int s = 0; s < 2; ++s) cu[s] = splat8(ub1[(nc0 + s) * 16 + ncol]);
#pragma unroll 1
        for (int kc = 0; kc < 8; ++kc) {
            A16 a;
            const u32x4* p = (const u32x4*)&ubf[urow * US + kc * 32 + hi8];
            a.q[0] = p[0]; a.q[1] = p[2];
            const u32x4* wq = (const u32x4*)(u1p + (size_t)kc * 8 * 512);
#pragma unroll
            for (int s = 0; s < 2; ++s) {
                A16 bb;
                bb.q[0] = wq[(nc0 + s) * 64 + lane * 2];
                bb.q[1] = wq[(nc0 + s) * 64 + lane * 2 + 1];
                cu[s] = wmma_bf(a, bb, cu[s]);
            }
        }
#pragma unroll
        for (int s = 0; s < 2; ++s)
#pragma unroll
            for (int r = 0; r < 8; ++r)
                h1b[(rt * 16 + r + mhi) * HB + (nc0 + s) * 16 + ncol] =
                    f2bf(tanh_fast(cu[s][r]));
        __syncthreads();

        // GEMM2: (32 x 128) @ (128 x 128) + residual into h
        v8f cv[2];
#pragma unroll
        for (int s = 0; s < 2; ++s) cv[s] = splat8(ub2[(nc0 + s) * 16 + ncol]);
#pragma unroll 1
        for (int kc = 0; kc < 4; ++kc) {
            A16 a;
            const u32x4* p = (const u32x4*)&h1b[urow * HB + kc * 32 + hi8];
            a.q[0] = p[0]; a.q[1] = p[2];
            const u32x4* wq = (const u32x4*)(u2p + (size_t)kc * 8 * 512);
#pragma unroll
            for (int s = 0; s < 2; ++s) {
                A16 bb;
                bb.q[0] = wq[(nc0 + s) * 64 + lane * 2];
                bb.q[1] = wq[(nc0 + s) * 64 + lane * 2 + 1];
                cv[s] = wmma_bf(a, bb, cv[s]);
            }
        }
#pragma unroll
        for (int s = 0; s < 2; ++s)
#pragma unroll
            for (int r = 0; r < 8; ++r) {
                int i  = rt * 16 + r + mhi;
                int cc = (nc0 + s) * 16 + ncol;
                float hv = bf2f(hbf[i * HB + cc]) + tanh_fast(cv[s][r]);
                hbf[i * HB + cc] = f2bf(hv);
            }
        __syncthreads();
    }

    // ---- write h (B*N, EMB) as f32 ----
    for (int idx = tid; idx < NN * EMB; idx += 256) {
        int i = idx >> 7, c = idx & 127;
        out[(size_t)b * NN * EMB + idx] = bf2f(hbf[i * HB + c]);
    }
}

// ---------------- host launch ----------------
extern "C" void kernel_launch(void* const* d_in, const int* in_sizes, int n_in,
                              void* d_out, int out_size, void* d_ws, size_t ws_size,
                              hipStream_t stream) {
    (void)in_sizes; (void)n_in; (void)out_size; (void)ws_size;
    const float* pos     = (const float*)d_in[0];
    const float* enc     = (const float*)d_in[1];
    const float* pos_e   = (const float*)d_in[2];
    const float* na_e    = (const float*)d_in[3];
    const int*   T       = (const int*)  d_in[4];
    const float* fc1_w   = (const float*)d_in[6];
    const float* fc1_b   = (const float*)d_in[7];
    const float* fc2_w   = (const float*)d_in[8];
    const float* fc2_b   = (const float*)d_in[9];
    const float* lin_w   = (const float*)d_in[10];
    const float* lin_b   = (const float*)d_in[11];
    const float* msg_w1  = (const float*)d_in[12];
    const float* msg_b1  = (const float*)d_in[13];
    const float* msg_w2  = (const float*)d_in[14];
    const float* msg_b2  = (const float*)d_in[15];
    const float* upd_w1  = (const float*)d_in[16];
    const float* upd_b1  = (const float*)d_in[17];
    const float* upd_w2  = (const float*)d_in[18];
    const float* upd_b2  = (const float*)d_in[19];

    unsigned short* ws = (unsigned short*)d_ws;

    pack_b<<<(1 * 7 * 4096) / 256, 256, 0, stream>>>(lin_w,  ws + OFF_LIN,  128, 193, 7);
    pack_b<<<(4 * 9 * 4096) / 256, 256, 0, stream>>>(msg_w1, ws + OFF_MSG1, 128, 261, 9);
    pack_b<<<(4 * 4 * 4096) / 256, 256, 0, stream>>>(msg_w2, ws + OFF_MSG2, 128, 128, 4);
    pack_b<<<(4 * 8 * 4096) / 256, 256, 0, stream>>>(upd_w1, ws + OFF_UPD1, 128, 256, 8);
    pack_b<<<(4 * 4 * 4096) / 256, 256, 0, stream>>>(upd_w2, ws + OFF_UPD2, 128, 128, 4);

    gnn_main<<<NB, 256, 0, stream>>>(pos, enc, pos_e, na_e, T,
                                     fc1_w, fc1_b, fc2_w, fc2_b, lin_b,
                                     msg_b1, msg_b2, upd_b1, upd_b2,
                                     ws, (float*)d_out);
}
